// HiResPrecipNet_61366492725290
// MI455X (gfx1250) — matrix operand order repaired and checked by reading, not compiled, not run on hardware
//
#include <hip/hip_runtime.h>
#include <hip/hip_bf16.h>
#include <cstdint>
#include <cstddef>

// ---------------------------------------------------------------------------
// Types for CDNA5 WMMA (wave32)
// ---------------------------------------------------------------------------
typedef __bf16 bf16;
typedef __attribute__((ext_vector_type(16))) __bf16 bf16x16;
typedef __attribute__((ext_vector_type(8)))  __bf16 bf16x8;
typedef __attribute__((ext_vector_type(8)))  float   f32x8;

#define THREADS 256

// ---------------------------------------------------------------------------
// Ordered-float encoding for atomicMax on float via unsigned atomics
// ---------------------------------------------------------------------------
__device__ __forceinline__ unsigned ford(float f) {
  unsigned u = __float_as_uint(f);
  return (u & 0x80000000u) ? ~u : (u | 0x80000000u);
}
__device__ __forceinline__ float funord(unsigned u) {
  return __uint_as_float((u & 0x80000000u) ? (u & 0x7fffffffu) : ~u);
}
#define FORD_NEG_INF 0x007FFFFFu   // ford(-inf)

// ---------------------------------------------------------------------------
// Fill kernels
// ---------------------------------------------------------------------------
__global__ void fill_f32(float* p, float v, long n) {
  long i = (long)blockIdx.x * blockDim.x + threadIdx.x;
  long s = (long)gridDim.x * blockDim.x;
  for (; i < n; i += s) p[i] = v;
}
__global__ void fill_u32(unsigned* p, unsigned v, long n) {
  long i = (long)blockIdx.x * blockDim.x + threadIdx.x;
  long s = (long)gridDim.x * blockDim.x;
  for (; i < n; i += s) p[i] = v;
}

// ---------------------------------------------------------------------------
// BatchNorm (training-mode, biased variance over node dim)
// ---------------------------------------------------------------------------
__global__ __launch_bounds__(THREADS)
void bn_stats(const float* __restrict__ X, int n, int F,
              float* __restrict__ mean, float* __restrict__ rstd) {
  __shared__ float s1[THREADS], s2[THREADS];
  int f = blockIdx.x;
  float a = 0.f, b = 0.f;
  for (int i = threadIdx.x; i < n; i += THREADS) {
    float v = X[(size_t)i * F + f];
    a += v; b += v * v;
  }
  s1[threadIdx.x] = a; s2[threadIdx.x] = b;
  __syncthreads();
  for (int s = THREADS / 2; s > 0; s >>= 1) {
    if ((int)threadIdx.x < s) {
      s1[threadIdx.x] += s1[threadIdx.x + s];
      s2[threadIdx.x] += s2[threadIdx.x + s];
    }
    __syncthreads();
  }
  if (threadIdx.x == 0) {
    float m  = s1[0] / (float)n;
    float va = s2[0] / (float)n - m * m;
    mean[f] = m;
    rstd[f] = rsqrtf(fmaxf(va, 0.f) + 1e-5f);
  }
}

__global__ void bn_apply(const float* __restrict__ X, float* __restrict__ Y,
                         const float* __restrict__ mean, const float* __restrict__ rstd,
                         const float* __restrict__ g, const float* __restrict__ b,
                         long total, int F, int relu) {
  long i = (long)blockIdx.x * blockDim.x + threadIdx.x;
  long s = (long)gridDim.x * blockDim.x;
  for (; i < total; i += s) {
    int f = (int)(i % F);
    float v = (X[i] - mean[f]) * rstd[f] * g[f] + b[f];
    if (relu) v = fmaxf(v, 0.f);
    Y[i] = v;
  }
}

// ---------------------------------------------------------------------------
// Operand packing for the WMMA GEMM (error-compensated bf16 hi/lo split).
// A: [M,K] fp32 -> Ah/Al [M,Kp] bf16 row-major, zero-padded (Kp = ceil32(K)).
// ---------------------------------------------------------------------------
__global__ void aconv(const float* __restrict__ A, bf16* __restrict__ Ah,
                      bf16* __restrict__ Al, long total, int K, int Kp) {
  long i = (long)blockIdx.x * blockDim.x + threadIdx.x;
  long st = (long)gridDim.x * blockDim.x;
  for (; i < total; i += st) {
    int k = (int)(i % Kp);
    long m = i / Kp;
    float v = (k < K) ? A[m * K + k] : 0.f;
    bf16 h = (bf16)v;
    Ah[i] = h;
    Al[i] = (bf16)(v - (float)h);
  }
}

// W: [K,N] fp32 -> fragment-order pack: Wp[(kb*N + n)*32 + j] = W[kb*32+j][n],
// so a lane's 16-element B fragment is 32 contiguous bytes.
__global__ void wpack(const float* __restrict__ W, bf16* __restrict__ Wh,
                      bf16* __restrict__ Wl, long total, int K, int N) {
  long i = (long)blockIdx.x * blockDim.x + threadIdx.x;
  long st = (long)gridDim.x * blockDim.x;
  for (; i < total; i += st) {
    int j = (int)(i & 31);
    long t = i >> 5;
    int n = (int)(t % N);
    long kb = t / N;
    int k = (int)(kb * 32 + j);
    float v = (k < K) ? W[(size_t)k * N + n] : 0.f;
    bf16 h = (bf16)v;
    Wh[i] = h;
    Wl[i] = (bf16)(v - (float)h);
  }
}

// ---------------------------------------------------------------------------
// GEMM: C[M,N] = A[M,K] @ W[K,N] (+bias), bf16x3 compensation with fp32 WMMA
// accumulate (Ah*Wh + Al*Wh + Ah*Wl). All operands pre-packed, no guards or
// conversions in the hot loop: 20 b128 loads + 12 v_wmma per k-step per wave.
// Block = 8 waves; wave computes 16 rows x 64 cols. N % 64 == 0, M % 16 == 0.
// ---------------------------------------------------------------------------
__global__ __launch_bounds__(THREADS)
void gemm_wmma(const bf16* __restrict__ Ah, const bf16* __restrict__ Al,
               const bf16* __restrict__ Wh, const bf16* __restrict__ Wl,
               const float* __restrict__ bias, float* __restrict__ C,
               int M, int N, int Kp) {
  const int lane = threadIdx.x & 31;
  const int wave = threadIdx.x >> 5;
  const int m0 = blockIdx.y * 128 + wave * 16;   // wave-uniform
  const int n0 = blockIdx.x * 64;
  if (m0 >= M) return;                           // whole wave exits together
  const int half = lane >> 4;                    // 0 or 1
  const int l16  = lane & 15;

  // A fragment chunks for this lane: [k0+16h, +16) contiguous
  const size_t rowA = (size_t)(m0 + l16) * Kp + 16 * half;
  f32x8 acc[4] = {};

  for (int k0 = 0; k0 < Kp; k0 += 32) {
    bf16x8 h0 = *(const bf16x8*)(Ah + rowA + k0);
    bf16x8 h1 = *(const bf16x8*)(Ah + rowA + k0 + 8);
    bf16x8 l0 = *(const bf16x8*)(Al + rowA + k0);
    bf16x8 l1 = *(const bf16x8*)(Al + rowA + k0 + 8);
    bf16x16 ah = __builtin_shufflevector(h0, h1, 0,1,2,3,4,5,6,7,8,9,10,11,12,13,14,15);
    bf16x16 al = __builtin_shufflevector(l0, l1, 0,1,2,3,4,5,6,7,8,9,10,11,12,13,14,15);

    const size_t wbase = ((size_t)(k0 >> 5) * N) * 32 + 16 * half;
#pragma unroll
    for (int t = 0; t < 4; ++t) {
      const size_t boff = wbase + (size_t)(n0 + 16 * t + l16) * 32;
      bf16x16 bh = *(const bf16x16*)(Wh + boff);
      bf16x16 bl = *(const bf16x16*)(Wl + boff);
      acc[t] = __builtin_amdgcn_wmma_f32_16x16x32_bf16(
                   false, ah, false, bh, (short)0, acc[t], false, false);
      acc[t] = __builtin_amdgcn_wmma_f32_16x16x32_bf16(
                   false, al, false, bh, (short)0, acc[t], false, false);
      acc[t] = __builtin_amdgcn_wmma_f32_16x16x32_bf16(
                   false, ah, false, bl, (short)0, acc[t], false, false);
    }
  }

  // ---- store (C/D layout: row = m0 + 8*half + r, col = n0t + l16)
#pragma unroll
  for (int t = 0; t < 4; ++t) {
    const int n = n0 + 16 * t + l16;
    const float bv = bias ? bias[n] : 0.f;
#pragma unroll
    for (int r = 0; r < 8; ++r) {
      int row = m0 + 8 * half + r;
      C[(size_t)row * N + n] = acc[t][r] + bv;
    }
  }
}

// ---------------------------------------------------------------------------
// GATv2 edge kernels. Wave per edge, lanes stride channels.
// ---------------------------------------------------------------------------
__global__ __launch_bounds__(THREADS)
void gat_logits(const float* __restrict__ xl, const float* __restrict__ xr,
                const int* __restrict__ src, const int* __restrict__ dst,
                const float* __restrict__ att, float* __restrict__ logit,
                unsigned* __restrict__ mord, float* __restrict__ cnt,
                int E, int H, int C) {
  int e = blockIdx.x * 8 + (threadIdx.x >> 5);
  int lane = threadIdx.x & 31;
  if (e >= E) return;
  int s = src[e], d = dst[e];
  const float* ps = xl + (size_t)s * H * C;
  const float* pd = xr ? xr + (size_t)d * H * C : nullptr;
  for (int h = 0; h < H; ++h) {
    float a = 0.f;
    for (int c = lane; c < C; c += 32) {
      float v = ps[h * C + c] + (pd ? pd[h * C + c] : 0.f);
      v = (v > 0.f) ? v : 0.2f * v;                 // leaky_relu 0.2
      a += v * att[h * C + c];
    }
    for (int off = 16; off; off >>= 1) a += __shfl_down(a, off, 32);
    if (lane == 0) {
      logit[(size_t)e * H + h] = a;
      atomicMax(&mord[(size_t)d * H + h], ford(a));
    }
  }
  if (lane == 0) atomicAdd(&cnt[d], 1.0f);
}

__global__ void gat_expsum(const float* __restrict__ logit, const int* __restrict__ dst,
                           const unsigned* __restrict__ mord, float* __restrict__ ew,
                           float* __restrict__ den, long total, int H) {
  long i = (long)blockIdx.x * blockDim.x + threadIdx.x;
  long st = (long)gridDim.x * blockDim.x;
  for (; i < total; i += st) {
    int e = (int)(i / H), h = (int)(i % H);
    int d = dst[e];
    float m = funord(mord[(size_t)d * H + h]);
    float w = expf(logit[i] - m);
    ew[i] = w;
    atomicAdd(&den[(size_t)d * H + h], w);
  }
}

__global__ __launch_bounds__(THREADS)
void gat_aggr(const float* __restrict__ xl, const int* __restrict__ src,
              const int* __restrict__ dst, const float* __restrict__ ew,
              const float* __restrict__ den, float* __restrict__ out,
              int E, int H, int C) {
  int e = blockIdx.x * 8 + (threadIdx.x >> 5);
  int lane = threadIdx.x & 31;
  if (e >= E) return;
  int s = src[e], d = dst[e];
  for (int h = 0; h < H; ++h) {
    float dn = den[(size_t)d * H + h];
    float alpha = ew[(size_t)e * H + h] / (dn > 0.f ? dn : 1.f);
    for (int c = lane; c < C; c += 32)
      atomicAdd(&out[((size_t)d * H + h) * C + c],
                xl[((size_t)s * H + h) * C + c] * alpha);
  }
}

__global__ void gat_fin(float* __restrict__ out, const float* __restrict__ cnt,
                        const float* __restrict__ bias, long total, int HC) {
  long i = (long)blockIdx.x * blockDim.x + threadIdx.x;
  long st = (long)gridDim.x * blockDim.x;
  for (; i < total; i += st) {
    int node = (int)(i / HC), j = (int)(i % HC);
    float c = fmaxf(cnt[node], 1.f);
    out[i] = out[i] / c + (bias ? bias[j] : 0.f);
  }
}

// ---------------------------------------------------------------------------
// Misc: concat [z_std | eh] -> [n,129]; wave-per-node dot for 128->1 heads
// ---------------------------------------------------------------------------
__global__ void concat_z(const float* __restrict__ zstd, const float* __restrict__ eh,
                         float* __restrict__ z, long n) {
  long i = (long)blockIdx.x * blockDim.x + threadIdx.x;
  long st = (long)gridDim.x * blockDim.x;
  long total = n * 129;
  for (; i < total; i += st) {
    long r = i / 129; int c = (int)(i % 129);
    z[i] = (c == 0) ? zstd[r] : eh[(size_t)r * 128 + (c - 1)];
  }
}

__global__ __launch_bounds__(THREADS)
void dotvec(const float* __restrict__ X, const float* __restrict__ w,
            const float* __restrict__ bptr, float* __restrict__ out, int n, int K) {
  int i = blockIdx.x * 8 + (threadIdx.x >> 5);
  int lane = threadIdx.x & 31;
  if (i >= n) return;
  float a = 0.f;
  for (int k = lane; k < K; k += 32) a += X[(size_t)i * K + k] * w[k];
  for (int off = 16; off; off >>= 1) a += __shfl_down(a, off, 32);
  if (lane == 0) out[i] = a + bptr[0];
}

// ---------------------------------------------------------------------------
// Host orchestration
// ---------------------------------------------------------------------------
static inline int gsb(long n) {            // grid size for stride kernels
  long b = (n + THREADS - 1) / THREADS;
  if (b > (1l << 20)) b = 1l << 20;
  if (b < 1) b = 1;
  return (int)b;
}

extern "C" void kernel_launch(void* const* d_in, const int* in_sizes, int n_in,
                              void* d_out, int out_size, void* d_ws, size_t ws_size,
                              hipStream_t stream) {
  const int NL = 30000, NH = 100000, F0 = 625;
  const int EL = 240000, E2 = 800000, EH = 800000;

  const float* x_low = (const float*)d_in[0];
  const float* z_std = (const float*)d_in[1];
  const int*   el    = (const int*)d_in[2];     // [2,EL]: src row0, dst row1
  const int*   l2hs  = (const int*)d_in[3];
  const int*   l2hd  = (const int*)d_in[4];
  const int*   ehg   = (const int*)d_in[5];     // [2,EH]
  const float* P[64];
  for (int i = 0; i < n_in && i < 64; ++i) P[i] = (const float*)d_in[i];
  // param indices (insertion order, None leaves dropped):
  // 6,7=bn0 g,b | 8..13=g1 Wl,att,bl,Wr,br,bias | 14,15=bn1 | 16..21=g2
  // 22,23=bn2 | 24..29=g3 | 30,31=l2h Wl,att | 32,33=bn3 | 34..39=h1
  // 40,41=bn4 | 42..47=h2 | 48,49=bn5 | 50..55=h3

  // ---- workspace bump allocator ----
  char* wp = (char*)d_ws;
  auto alloc = [&](size_t bytes) -> void* {
    void* p = (void*)wp;
    wp += (bytes + 255) & ~(size_t)255;
    return p;
  };
  float*    A0   = (float*)alloc((size_t)NL * F0 * 4);        // 75 MB (reused)
  float*    XL   = (float*)alloc((size_t)NH * 256 * 4);       // 102.4 MB
  float*    XR   = (float*)alloc((size_t)NH * 256 * 4);       // 102.4 MB
  float*    OUTB = (float*)alloc((size_t)NH * 256 * 4);       // 102.4 MB
  float*    Z    = (float*)alloc((size_t)NH * 129 * 4);       // 51.6 MB
  bf16*     Ahp  = (bf16*)alloc((size_t)NH * 256 * 2);        // 51.2 MB
  bf16*     Alp  = (bf16*)alloc((size_t)NH * 256 * 2);        // 51.2 MB
  bf16*     Whb  = (bf16*)alloc((size_t)640 * 512 * 2);
  bf16*     Wlb  = (bf16*)alloc((size_t)640 * 512 * 2);
  float*    logit= (float*)alloc((size_t)EH * 2 * 4);
  float*    ew   = (float*)alloc((size_t)EH * 2 * 4);
  unsigned* mord = (unsigned*)alloc((size_t)NH * 2 * 4);
  float*    den  = (float*)alloc((size_t)NH * 2 * 4);
  float*    cnt  = (float*)alloc((size_t)NH * 4);
  float*    xls  = (float*)alloc((size_t)NH * 4);
  float*    xrs  = (float*)alloc((size_t)NH * 4);
  float*    bnm  = (float*)alloc(1024 * 4);
  float*    bnr  = (float*)alloc(1024 * 4);

  auto bn = [&](const float* X, float* Y, const float* g, const float* b,
                int n, int F, int relu) {
    bn_stats<<<F, THREADS, 0, stream>>>(X, n, F, bnm, bnr);
    long tot = (long)n * F;
    bn_apply<<<gsb(tot), THREADS, 0, stream>>>(X, Y, bnm, bnr, g, b, tot, F, relu);
  };

  // One A-split per layer, then 1-2 GEMMs sharing it.
  auto gemmPair = [&](const float* A, int M, int K,
                      const float* W1, const float* b1, float* C1,
                      const float* W2, const float* b2, float* C2, int N) {
    int Kp = (K + 31) & ~31;
    long at = (long)M * Kp;
    aconv<<<gsb(at), THREADS, 0, stream>>>(A, Ahp, Alp, at, K, Kp);
    long wt = (long)Kp * N;
    dim3 g(N / 64, (M + 127) / 128);
    wpack<<<gsb(wt), THREADS, 0, stream>>>(W1, Whb, Wlb, wt, K, N);
    gemm_wmma<<<g, THREADS, 0, stream>>>(Ahp, Alp, Whb, Wlb, b1, C1, M, N, Kp);
    if (W2) {
      wpack<<<gsb(wt), THREADS, 0, stream>>>(W2, Whb, Wlb, wt, K, N);
      gemm_wmma<<<g, THREADS, 0, stream>>>(Ahp, Alp, Whb, Wlb, b2, C2, M, N, Kp);
    }
  };

  auto gat = [&](const float* xl, const float* xr, const int* src, const int* dst,
                 const float* att, const float* bias, float* out,
                 int n_dst, int E, int H, int C) {
    long nh = (long)n_dst * H;
    fill_u32<<<gsb(nh), THREADS, 0, stream>>>(mord, FORD_NEG_INF, nh);
    fill_f32<<<gsb(nh), THREADS, 0, stream>>>(den, 0.f, nh);
    fill_f32<<<gsb(n_dst), THREADS, 0, stream>>>(cnt, 0.f, (long)n_dst);
    long ohc = (long)n_dst * H * C;
    fill_f32<<<gsb(ohc), THREADS, 0, stream>>>(out, 0.f, ohc);
    int eb = (E + 7) / 8;
    gat_logits<<<eb, THREADS, 0, stream>>>(xl, xr, src, dst, att, logit, mord, cnt, E, H, C);
    long ehh = (long)E * H;
    gat_expsum<<<gsb(ehh), THREADS, 0, stream>>>(logit, dst, mord, ew, den, ehh, H);
    gat_aggr<<<eb, THREADS, 0, stream>>>(xl, src, dst, ew, den, out, E, H, C);
    gat_fin<<<gsb(ohc), THREADS, 0, stream>>>(out, cnt, bias, ohc, H * C);
  };

  // ================= low_net =================
  bn(x_low, A0, P[6], P[7], NL, F0, 0);                       // bn0
  gemmPair(A0, NL, F0, P[8], P[10], XL, P[11], P[12], XR, 512);   // g1
  gat(XL, XR, el, el + EL, P[9], P[13], OUTB, NL, EL, 1, 512);
  bn(OUTB, OUTB, P[14], P[15], NL, 512, 1);                   // bn1+relu
  gemmPair(OUTB, NL, 512, P[16], P[18], XL, P[19], P[20], XR, 512); // g2
  gat(XL, XR, el, el + EL, P[17], P[21], A0, NL, EL, 1, 512);
  bn(A0, A0, P[22], P[23], NL, 512, 1);                       // bn2+relu
  gemmPair(A0, NL, 512, P[24], P[26], XL, P[27], P[28], XR, 512);  // g3
  gat(XL, XR, el, el + EL, P[25], P[29], OUTB, NL, EL, 1, 512);

  // ================= low2high (bipartite, no dst branch, no bias) =========
  gemmPair(OUTB, NL, 512, P[30], nullptr, XL, nullptr, nullptr, nullptr, 128);
  gat(XL, nullptr, l2hs, l2hd, P[31], nullptr, A0 /*eh*/, NH, E2, 1, 128);

  // ================= high_net =================
  concat_z<<<gsb((long)NH * 129), THREADS, 0, stream>>>(z_std, A0, Z, NH);
  bn(Z, Z, P[32], P[33], NH, 129, 0);                         // bn3
  gemmPair(Z, NH, 129, P[34], P[36], XL, P[37], P[38], XR, 256);   // h1 (H=2,C=128)
  gat(XL, XR, ehg, ehg + EH, P[35], P[39], OUTB, NH, EH, 2, 128);
  bn(OUTB, OUTB, P[40], P[41], NH, 256, 1);                   // bn4+relu
  gemmPair(OUTB, NH, 256, P[42], P[44], XL, P[45], P[46], XR, 128); // h2
  gat(XL, XR, ehg, ehg + EH, P[43], P[47], A0, NH, EH, 1, 128);
  bn(A0, A0, P[48], P[49], NH, 128, 1);                       // bn5+relu
  // h3: 128 -> 1 per node, then scalar GAT into d_out
  dotvec<<<(NH + 7) / 8, THREADS, 0, stream>>>(A0, P[50], P[52], xls, NH, 128);
  dotvec<<<(NH + 7) / 8, THREADS, 0, stream>>>(A0, P[53], P[54], xrs, NH, 128);
  gat(xls, xrs, ehg, ehg + EH, P[51], P[55], (float*)d_out, NH, EH, 1, 1);

  (void)in_sizes; (void)n_in; (void)out_size; (void)ws_size;
}
